// HomeostaticEqProp_46084999086212
// MI455X (gfx1250) — matrix operand order repaired and checked by reading, not compiled, not run on hardware
//
#include <hip/hip_runtime.h>

// ---------------------------------------------------------------------------
// HomeostaticEqProp on MI455X (gfx1250, wave32, WMMA).
//
// Each 256-thread block owns a 16-row batch tile for the whole 30-step
// recurrence. Hidden state h[5][16][256] lives in fp32 VGPR accumulators
// (10 tiles/wave x v8f). A bf16 snapshot of the previous state ("pres")
// lives in LDS, rebuilt once per step. Weights are pre-converted to bf16
// in d_ws (800KB -> L2 resident) and streamed as WMMA B-fragments
// (contiguous 32B/lane; W is [out,in] row-major so no transpose needed).
// HBM traffic ~ read x (67MB) + write out (17MB) only.
//
// This revision: the two column tiles a wave owns per layer share one
// A-fragment explicitly (one LDS A load feeds two back-to-back WMMAs),
// halving ds traffic in the hot loop; same sharing in the x_emb phase.
// ---------------------------------------------------------------------------

typedef __attribute__((ext_vector_type(16))) __bf16 bf16x16;
typedef __attribute__((ext_vector_type(8)))  __bf16 bf16x8;
typedef __attribute__((ext_vector_type(8)))  float  f32x8;

#define BATCH      65536
#define IN_DIM     256
#define HID        256
#define OUT_DIM    64
#define NUM_LAYERS 5
#define BT         16          // batch rows per block
#define PRES_STRIDE 264        // bf16 row stride (256 + 8 pad -> 4-bank skew)

// d_ws layout (bf16 elements):
#define WS_WIN   0                                   // W_in   [256 x 256]
#define WS_W     (IN_DIM * HID)                      // W*scale[5 x 256 x 256]
#define WS_HEAD  (WS_W + NUM_LAYERS * HID * HID)     // head_w [64 x 256]
#define WS_TOTAL (WS_HEAD + OUT_DIM * HID)

// ---------------------------------------------------------------------------
// Pre-pass: convert weights to bf16 (W scaled by per-layer scale).
// ---------------------------------------------------------------------------
__global__ void prep_weights_k(const float* __restrict__ win,
                               const float* __restrict__ W,
                               const float* __restrict__ scales,
                               const float* __restrict__ headw,
                               __bf16* __restrict__ ws) {
  int i = blockIdx.x * blockDim.x + threadIdx.x;
  if (i >= WS_TOTAL) return;
  float v;
  if (i < WS_W) {
    v = win[i];
  } else if (i < WS_HEAD) {
    int j = i - WS_W;
    v = W[j] * scales[j >> 16];          // 65536 elems per layer
  } else {
    v = headw[i - WS_HEAD];
  }
  ws[i] = (__bf16)v;
}

// A-fragment assembly from two 8-element chunks (contiguous k order used here:
// lane<16 -> K {0..15}, lane>=16 -> K {16..31} rel. to k0).
__device__ __forceinline__ bf16x16 make_a(bf16x8 lo, bf16x8 hi) {
  bf16x16 a;
#pragma unroll
  for (int i = 0; i < 8; ++i) { a[i] = lo[i]; a[i + 8] = hi[i]; }
  return a;
}

// B-fragment: lane<16 -> col N=n0+lane, K=k0..k0+15 contiguous;
// lanes 16..31 -> same col, K=k0+16..k0+31. Row-major [out,in] weights
// make this a single contiguous 32B load per lane.
__device__ __forceinline__ bf16x16 load_b(const __bf16* w, int row, int k) {
  return *(const bf16x16*)(w + (size_t)row * HID + k);
}

// ---------------------------------------------------------------------------
// Main persistent-tile kernel.
// ---------------------------------------------------------------------------
__launch_bounds__(256, 1)
__global__ void eqprop_main_k(const float* __restrict__ x,
                              const float* __restrict__ win_b,
                              const float* __restrict__ b,
                              const float* __restrict__ head_b,
                              const __bf16* __restrict__ ws,
                              const int* __restrict__ stepsp,
                              float* __restrict__ out) {
  __shared__ __bf16 pres[NUM_LAYERS * BT * PRES_STRIDE];   // 42,240 B

  const int tid  = threadIdx.x;
  const int wave = tid >> 5;
  const int lane = tid & 31;
  const int m    = lane & 15;     // row (A) / column (B,D) index within tile
  const int half = lane >> 4;
  const int b0   = blockIdx.x * BT;
  const int steps = *stepsp;

  const int n0A = wave << 4;        // first owned column tile
  const int n0B = (8 + wave) << 4;  // second owned column tile

  // ---- zero pres (initial hidden state is zero) ----
  for (int i = tid; i < NUM_LAYERS * BT * PRES_STRIDE; i += 256)
    pres[i] = (__bf16)0.0f;
  __syncthreads();

  // ---- x_emb = x @ W_in^T + b_in  -> pres[0] (bf16, step-invariant) ----
  {
    f32x8 dA = {}, dB = {};
#pragma unroll
    for (int k0 = 0; k0 < IN_DIM; k0 += 32) {
      const float* xp = x + (size_t)(b0 + m) * IN_DIM + k0 + half * 16;
      bf16x16 a;
#pragma unroll
      for (int i = 0; i < 8; ++i) {
        a[i]     = (__bf16)xp[i];
        a[i + 8] = (__bf16)xp[i + 8];
      }
      const __bf16* wl = ws + WS_WIN;
      bf16x16 bmA = load_b(wl, n0A + m, k0 + (half << 4));
      bf16x16 bmB = load_b(wl, n0B + m, k0 + (half << 4));
      dA = __builtin_amdgcn_wmma_f32_16x16x32_bf16(false, a, false, bmA,
                                                   (short)0, dA, false, false);
      dB = __builtin_amdgcn_wmma_f32_16x16x32_bf16(false, a, false, bmB,
                                                   (short)0, dB, false, false);
    }
    const float biasA = win_b[n0A + m];
    const float biasB = win_b[n0B + m];
#pragma unroll
    for (int r = 0; r < 8; ++r) {
      const int row = (r + 8 * half) * PRES_STRIDE;
      pres[row + n0A + m] = (__bf16)(dA[r] + biasA);
      pres[row + n0B + m] = (__bf16)(dB[r] + biasB);
    }
  }
  __syncthreads();

  // ---- per-wave tile set: j -> (layer l = j>>1, cols n0A / n0B) ----
  f32x8 h[10];
  float biasv[10];
#pragma unroll
  for (int j = 0; j < 10; ++j) {
    h[j] = (f32x8){};
    const int l  = j >> 1;
    const int n0 = (j & 1) ? n0B : n0A;
    biasv[j] = b[l * HID + n0 + m];
  }

  // ---- 30 fixed-point steps ----
  for (int s = 0; s < steps; ++s) {
    // publish old state: pres[l+1] <- bf16(h[l])  (layer 4 feeds nothing)
#pragma unroll
    for (int j = 0; j < 8; ++j) {                 // j=0..7 -> l=0..3
      const int l  = j >> 1;
      const int n0 = (j & 1) ? n0B : n0A;
      __bf16* p = pres + (l + 1) * BT * PRES_STRIDE + n0 + m;
#pragma unroll
      for (int r = 0; r < 8; ++r)
        p[(r + 8 * half) * PRES_STRIDE] = (__bf16)h[j][r];
    }
    __syncthreads();

    // compute: h[l] = 0.5*h[l] + 0.5*tanh(pres[l] @ W_l^T + b_l)
    // One shared A-fragment per (l, k0) feeds both owned column tiles.
#pragma unroll
    for (int l = 0; l < NUM_LAYERS; ++l) {
      const int jA = 2 * l, jB = 2 * l + 1;
      const __bf16* pl = pres + l * BT * PRES_STRIDE;
      const __bf16* wl = ws + WS_W + (size_t)l * HID * HID;
      f32x8 dA = {}, dB = {};
#pragma unroll
      for (int k0 = 0; k0 < HID; k0 += 32) {
        const __bf16* ap = pl + m * PRES_STRIDE + k0 + half * 16;
        bf16x16 a   = make_a(*(const bf16x8*)ap, *(const bf16x8*)(ap + 8));
        bf16x16 bmA = load_b(wl, n0A + m, k0 + (half << 4));
        bf16x16 bmB = load_b(wl, n0B + m, k0 + (half << 4));
        dA = __builtin_amdgcn_wmma_f32_16x16x32_bf16(false, a, false, bmA,
                                                     (short)0, dA, false, false);
        dB = __builtin_amdgcn_wmma_f32_16x16x32_bf16(false, a, false, bmB,
                                                     (short)0, dB, false, false);
      }
#pragma unroll
      for (int r = 0; r < 8; ++r) {
        h[jA][r] = 0.5f * h[jA][r] + 0.5f * tanhf(dA[r] + biasv[jA]);
        h[jB][r] = 0.5f * h[jB][r] + 0.5f * tanhf(dB[r] + biasv[jB]);
      }
    }
    __syncthreads();
  }

  // ---- head: out = h[4] @ head_w^T + head_b ----
  // publish h[4] (j=8,9) into pres[0] region (x_emb no longer needed)
#pragma unroll
  for (int j = 8; j < 10; ++j) {
    const int n0 = (j & 1) ? n0B : n0A;
    __bf16* p = pres + n0 + m;
#pragma unroll
    for (int r = 0; r < 8; ++r)
      p[(r + 8 * half) * PRES_STRIDE] = (__bf16)h[j][r];
  }
  __syncthreads();

  if (wave < 4) {                         // 4 output column tiles (OUT=64)
    const int n0 = wave << 4;
    const __bf16* hw = ws + WS_HEAD;
    f32x8 d = {};
#pragma unroll
    for (int k0 = 0; k0 < HID; k0 += 32) {
      const __bf16* ap = pres + m * PRES_STRIDE + k0 + half * 16;
      bf16x16 a  = make_a(*(const bf16x8*)ap, *(const bf16x8*)(ap + 8));
      bf16x16 bm = load_b(hw, n0 + m, k0 + (half << 4));
      d = __builtin_amdgcn_wmma_f32_16x16x32_bf16(false, a, false, bm,
                                                  (short)0, d, false, false);
    }
    const float hb = head_b[n0 + m];
#pragma unroll
    for (int r = 0; r < 8; ++r)
      out[(size_t)(b0 + r + 8 * half) * OUT_DIM + n0 + m] = d[r] + hb;
  }
}

// ---------------------------------------------------------------------------
// Host entry
// ---------------------------------------------------------------------------
extern "C" void kernel_launch(void* const* d_in, const int* in_sizes, int n_in,
                              void* d_out, int out_size, void* d_ws, size_t ws_size,
                              hipStream_t stream) {
  const float* x      = (const float*)d_in[0];
  const float* winw   = (const float*)d_in[1];
  const float* winb   = (const float*)d_in[2];
  const float* W      = (const float*)d_in[3];
  const float* bb     = (const float*)d_in[4];
  const float* scales = (const float*)d_in[5];
  const float* headw  = (const float*)d_in[6];
  const float* headb  = (const float*)d_in[7];
  const int*   steps  = (const int*)d_in[8];
  __bf16* ws  = (__bf16*)d_ws;
  float*  out = (float*)d_out;

  prep_weights_k<<<(WS_TOTAL + 255) / 256, 256, 0, stream>>>(
      winw, W, scales, headw, ws);

  eqprop_main_k<<<BATCH / BT, 256, 0, stream>>>(
      x, winb, bb, headb, ws, steps, out);
}
